// EdgeNet_29360396436029
// MI455X (gfx1250) — matrix operand order repaired and checked by reading, not compiled, not run on hardware
//
#include <hip/hip_runtime.h>
#include <hip/hip_bf16.h>

typedef __attribute__((ext_vector_type(16))) _Float16 v16h;
typedef __attribute__((ext_vector_type(8)))  _Float16 v8h;
typedef __attribute__((ext_vector_type(8)))  float    v8f;

#define LDK 72   // LDS row stride in halves (144 B -> 16B-aligned segments)

// ---------------------------------------------------------------------------
// WMMA helper (CDNA5 16x16x32 f16 -> f32)
// ---------------------------------------------------------------------------
__device__ __forceinline__ v8f wmma_f16(v16h a, v16h b, v8f c) {
  // 8 args: (neg_a, A, neg_b, B, c_mod, C, reuse_a, reuse_b)
  return __builtin_amdgcn_wmma_f32_16x16x32_f16(false, a, false, b,
                                                (short)0, c, false, false);
}

__device__ __forceinline__ v16h cat8(v8h lo, v8h hi) {
  return __builtin_shufflevector(lo, hi, 0, 1, 2, 3, 4, 5, 6, 7,
                                 8, 9, 10, 11, 12, 13, 14, 15);
}

// A-fragment (16xK f16) from per-wave LDS tile [16][LDK].
// lane = m + 16*hi ; halves 0..15 -> k = kOff+hi*16+0..15.
// Both 8-half runs are contiguous & 16B aligned -> two ds_load_b128.
__device__ __forceinline__ v16h loadA_tile(const _Float16* tile, int lane, int kOff) {
  const int m = lane & 15, hi = lane >> 4;
  const _Float16* p = tile + m * LDK + kOff + hi * 16;
  const v8h lo = *(const v8h*)p;
  const v8h hh = *(const v8h*)(p + 8);
  return cat8(lo, hh);
}

// B-fragment from pre-swizzled weight block: wblk[f*512 + lane*16 + h]
// -> two global_load_b128 per fragment.
__device__ __forceinline__ v16h loadB_frag(const _Float16* __restrict__ W,
                                           int lane, int f) {
  const _Float16* p = W + f * 512 + lane * 16;
  const v8h lo = *(const v8h*)p;
  const v8h hh = *(const v8h*)(p + 8);
  return cat8(lo, hh);
}

// D-fragment (16x16 f32) -> bias + optional ReLU -> f16 tile
// component r, lane L -> element (M = r + 8*(L>>4), N = L&15)
__device__ __forceinline__ void storeD_tile(_Float16* tile, int lane, int nOff,
                                            v8f acc, float bias, bool relu) {
  const int n = lane & 15, hi = lane >> 4;
#pragma unroll
  for (int r = 0; r < 8; ++r) {
    float v = acc[r] + bias;
    if (relu) v = fmaxf(v, 0.0f);
    tile[(hi * 8 + r) * LDK + nOff + n] = (_Float16)v;
  }
}

// ---------------------------------------------------------------------------
// EdgeConv: per-edge MLP via WMMA, 16 edges per wave, atomic sum at dst.
// Weight block = 11 pre-swizzled B fragments:
//   f0,f1   : layer1 (K=32 padded, N tiles 0,16)
//   f2..f5  : layer2 (K=32, N tiles 0,16,32,48)
//   f6..f9  : layer3 (N tile 0: K 0/32 ; N tile 16: K 0/32)
//   f10     : layer4 (K=32, N=16 padded)
// Bias block: b1[32] b2[64] b3[32] b4[16]
// ---------------------------------------------------------------------------
template <int INF, int OUTF, bool LAST_RELU>
__global__ __launch_bounds__(256) void edgeconv_wmma(
    const float* __restrict__ xin,
    const int* __restrict__ src, const int* __restrict__ dst,
    const _Float16* __restrict__ wblk, const float* __restrict__ bblk,
    float* __restrict__ accum, int nEdges) {
  constexpr int XD = INF / 2;

  __shared__ _Float16 tiles[8][16 * LDK];
  __shared__ float    ostage[8][16 * 4];

  const int lane = threadIdx.x & 31;
  const int wv   = threadIdx.x >> 5;
  _Float16* tile = &tiles[wv][0];
  float*    ost  = &ostage[wv][0];

  const int waveGlobal = blockIdx.x * 8 + wv;
  const int nWaves     = gridDim.x * 8;
  const int nGroups    = (nEdges + 15) >> 4;

  // Weight fragments held in VGPRs for the whole wave lifetime.
  v16h B1[2], B2[4], B3[4], B4f;
#pragma unroll
  for (int t = 0; t < 2; ++t) B1[t] = loadB_frag(wblk, lane, t);
#pragma unroll
  for (int t = 0; t < 4; ++t) B2[t] = loadB_frag(wblk, lane, 2 + t);
#pragma unroll
  for (int t = 0; t < 4; ++t) B3[t] = loadB_frag(wblk, lane, 6 + t);
  B4f = loadB_frag(wblk, lane, 10);

  const int n  = lane & 15;
  const int hi = lane >> 4;
  const float* b1 = bblk;
  const float* b2 = bblk + 32;
  const float* b3 = bblk + 96;
  const float* b4 = bblk + 128;
  const float bb1[2] = {b1[n], b1[16 + n]};
  const float bb2[4] = {b2[n], b2[16 + n], b2[32 + n], b2[48 + n]};
  const float bb3[2] = {b3[n], b3[16 + n]};
  const float bb4    = b4[n];

  for (int g = waveGlobal; g < nGroups; g += nWaves) {
    const int e0 = g << 4;

    // ---- Build A1 (16 edges x INF features, zero-padded to K=32) ----
    v16h A;
#pragma unroll
    for (int h = 0; h < 16; ++h) A[h] = (_Float16)0.0f;
    {
      const int e = e0 + n;
      if (hi == 0 && e < nEdges) {   // hi==0 lanes own k = 0..7
        const int di = dst[e], si = src[e];
        if constexpr (XD == 4) {
          const float4 xi = ((const float4*)xin)[di];
          const float4 xj = ((const float4*)xin)[si];
          A[0] = (_Float16)xi.x; A[1] = (_Float16)xi.y;
          A[2] = (_Float16)xi.z; A[3] = (_Float16)xi.w;
          A[4] = (_Float16)(xj.x - xi.x); A[5] = (_Float16)(xj.y - xi.y);
          A[6] = (_Float16)(xj.z - xi.z); A[7] = (_Float16)(xj.w - xi.w);
        } else {
          const float2 xi = ((const float2*)xin)[di];
          const float2 xj = ((const float2*)xin)[si];
          A[0] = (_Float16)xi.x; A[1] = (_Float16)xi.y;
          A[2] = (_Float16)(xj.x - xi.x); A[3] = (_Float16)(xj.y - xi.y);
        }
      }
    }

    // ---- Layer 1: [16x32] @ [32x32] ----
#pragma unroll
    for (int t = 0; t < 2; ++t) {
      v8f c = {};
      c = wmma_f16(A, B1[t], c);
      storeD_tile(tile, lane, 16 * t, c, bb1[t], true);
    }

    // ---- Layer 2: [16x32] @ [32x64] ----
    {
      const v16h A2 = loadA_tile(tile, lane, 0);
#pragma unroll
      for (int t = 0; t < 4; ++t) {
        v8f c = {};
        c = wmma_f16(A2, B2[t], c);
        storeD_tile(tile, lane, 16 * t, c, bb2[t], true);
      }
    }

    // ---- Layer 3: [16x64] @ [64x32] (K chained as 2x32) ----
    {
      const v16h A3a = loadA_tile(tile, lane, 0);
      const v16h A3b = loadA_tile(tile, lane, 32);
#pragma unroll
      for (int t = 0; t < 2; ++t) {
        v8f c = {};
        c = wmma_f16(A3a, B3[2 * t + 0], c);
        c = wmma_f16(A3b, B3[2 * t + 1], c);
        storeD_tile(tile, lane, 16 * t, c, bb3[t], true);
      }
    }

    // ---- Layer 4: [16x32] @ [32x16] (only OUTF cols valid) ----
    {
      const v16h A4 = loadA_tile(tile, lane, 0);
      v8f c = {};
      c = wmma_f16(A4, B4f, c);
      if (n < OUTF) {
#pragma unroll
        for (int r = 0; r < 8; ++r) {
          float v = c[r] + bb4;
          if (LAST_RELU) v = fmaxf(v, 0.0f);
          ost[(hi * 8 + r) * OUTF + n] = v;
        }
      }
      // Scatter-add messages at destination nodes (per-wave LDS, in-order).
      for (int idx = lane; idx < 16 * OUTF; idx += 32) {
        const int mm = idx / OUTF, ff = idx - mm * OUTF;
        const int ee = e0 + mm;
        if (ee < nEdges) {
          atomicAdd(&accum[dst[ee] * OUTF + ff], ost[idx]);
        }
      }
    }
  }
}

// ---------------------------------------------------------------------------
// Small helper kernels
// ---------------------------------------------------------------------------
__global__ void zero_f32(float* p, int nelem) {
  int i = blockIdx.x * blockDim.x + threadIdx.x;
  int stride = gridDim.x * blockDim.x;
  for (; i < nelem; i += stride) p[i] = 0.0f;
}

__global__ void count_edges(const int* __restrict__ dst, float* __restrict__ cnt,
                            int nEdges) {
  int i = blockIdx.x * blockDim.x + threadIdx.x;
  int stride = gridDim.x * blockDim.x;
  for (; i < nEdges; i += stride) atomicAdd(&cnt[dst[i]], 1.0f);
}

__global__ void mean_div(const float* __restrict__ acc, const float* __restrict__ cnt,
                         float* __restrict__ out, int nNodes, int feats) {
  int i = blockIdx.x * blockDim.x + threadIdx.x;
  int stride = gridDim.x * blockDim.x;
  const int total = nNodes * feats;
  for (; i < total; i += stride) {
    const int node = i / feats;
    out[i] = acc[i] / fmaxf(cnt[node], 1.0f);
  }
}

// Build pre-swizzled f16 weight fragments (11 frags x 32 lanes x 16 halves)
// + bias block b1[32] b2[64] b3[32] b4[16].
// Fragment f, lane L, half h holds W_layer[k][n] with
//   k = kOff[f] + (L>>4)*16 + h ,  n = nOff[f] + (L&15)   (zero-padded).
__global__ void prep_block(const float* __restrict__ w1, const float* __restrict__ w2,
                           const float* __restrict__ w3, const float* __restrict__ w4,
                           const float* __restrict__ b1, const float* __restrict__ b2,
                           const float* __restrict__ b3, const float* __restrict__ b4,
                           int inf, int outf,
                           _Float16* __restrict__ wblk, float* __restrict__ bblk) {
  const int t0 = blockIdx.x * blockDim.x + threadIdx.x;
  const int stride = gridDim.x * blockDim.x;

  const int layerOf[11] = {1, 1, 2, 2, 2, 2, 3, 3, 3, 3, 4};
  const int kOff[11]    = {0, 0, 0, 0, 0, 0, 0, 32, 0, 32, 0};
  const int nOff[11]    = {0, 16, 0, 16, 32, 48, 0, 0, 16, 16, 0};

  for (int i = t0; i < 11 * 512; i += stride) {
    const int f    = i >> 9;
    const int r    = i & 511;
    const int lane = r >> 4;
    const int h    = r & 15;
    const int k    = kOff[f] + (lane >> 4) * 16 + h;
    const int nn   = nOff[f] + (lane & 15);
    float v = 0.0f;
    switch (layerOf[f]) {
      case 1: if (k < inf)  v = w1[k * 32 + nn];   break;  // [inf x 32]
      case 2:               v = w2[k * 64 + nn];   break;  // [32 x 64]
      case 3:               v = w3[k * 32 + nn];   break;  // [64 x 32]
      case 4: if (nn < outf) v = w4[k * outf + nn]; break; // [32 x outf]
    }
    wblk[i] = (_Float16)v;
  }
  for (int i = t0; i < 32; i += stride) bblk[i] = b1[i];
  for (int i = t0; i < 64; i += stride) bblk[32 + i] = b2[i];
  for (int i = t0; i < 32; i += stride) bblk[96 + i] = b3[i];
  for (int i = t0; i < 16; i += stride) bblk[128 + i] = (i < outf) ? b4[i] : 0.0f;
}

// ---------------------------------------------------------------------------
// Host launcher
// ---------------------------------------------------------------------------
extern "C" void kernel_launch(void* const* d_in, const int* in_sizes, int n_in,
                              void* d_out, int out_size, void* d_ws, size_t ws_size,
                              hipStream_t stream) {
  const int nNodes = in_sizes[0] / 4;   // x: [N,4] f32
  const int nEdges = in_sizes[1] / 2;   // edge_index: [2,E]

  const float* x   = (const float*)d_in[0];
  const int*   src = (const int*)d_in[1];
  const int*   dst = src + nEdges;

  const float* enc_w[4] = {(const float*)d_in[2], (const float*)d_in[4],
                           (const float*)d_in[6], (const float*)d_in[8]};
  const float* enc_b[4] = {(const float*)d_in[3], (const float*)d_in[5],
                           (const float*)d_in[7], (const float*)d_in[9]};
  const float* dec_w[4] = {(const float*)d_in[10], (const float*)d_in[12],
                           (const float*)d_in[14], (const float*)d_in[16]};
  const float* dec_b[4] = {(const float*)d_in[11], (const float*)d_in[13],
                           (const float*)d_in[15], (const float*)d_in[17]};

  // Workspace carving (256B aligned regions)
  size_t off = 0;
  auto carve = [&](size_t bytes) -> void* {
    void* p = (char*)d_ws + off;
    off += (bytes + 255) & ~(size_t)255;
    return p;
  };
  float*     cnt    = (float*)carve((size_t)nNodes * 4);
  float*     zsum   = (float*)carve((size_t)nNodes * 2 * 4);
  float*     outsum = (float*)carve((size_t)nNodes * 4 * 4);
  float*     z      = (float*)carve((size_t)nNodes * 2 * 4);
  _Float16*  wEnc   = (_Float16*)carve(11 * 512 * 2);
  float*     bEnc   = (float*)carve(144 * 4);
  _Float16*  wDec   = (_Float16*)carve(11 * 512 * 2);
  float*     bDec   = (float*)carve(144 * 4);
  (void)ws_size;

  // 1) Prepare pre-swizzled f16 weight fragments
  prep_block<<<8, 256, 0, stream>>>(enc_w[0], enc_w[1], enc_w[2], enc_w[3],
                                    enc_b[0], enc_b[1], enc_b[2], enc_b[3],
                                    8, 2, wEnc, bEnc);
  prep_block<<<8, 256, 0, stream>>>(dec_w[0], dec_w[1], dec_w[2], dec_w[3],
                                    dec_b[0], dec_b[1], dec_b[2], dec_b[3],
                                    4, 4, wDec, bDec);

  // 2) Zero accumulators + counts
  zero_f32<<<(nNodes + 255) / 256, 256, 0, stream>>>(cnt, nNodes);
  zero_f32<<<(nNodes * 2 + 255) / 256, 256, 0, stream>>>(zsum, nNodes * 2);
  zero_f32<<<(nNodes * 4 + 255) / 256, 256, 0, stream>>>(outsum, nNodes * 4);

  // 3) Per-destination edge counts (shared by both EdgeConv layers)
  count_edges<<<(nEdges + 255) / 256, 256, 0, stream>>>(dst, cnt, nEdges);

  // 4) Encoder EdgeConv (8 -> 32 -> 64 -> 32 -> 2, ReLU incl. last)
  edgeconv_wmma<8, 2, true><<<2048, 256, 0, stream>>>(x, src, dst, wEnc, bEnc,
                                                      zsum, nEdges);
  mean_div<<<(nNodes * 2 + 255) / 256, 256, 0, stream>>>(zsum, cnt, z, nNodes, 2);

  // 5) Decoder EdgeConv (4 -> 32 -> 64 -> 32 -> 4, no final ReLU)
  edgeconv_wmma<4, 4, false><<<2048, 256, 0, stream>>>(z, src, dst, wDec, bDec,
                                                       outsum, nEdges);
  mean_div<<<(nNodes * 4 + 255) / 256, 256, 0, stream>>>(outsum, cnt,
                                                         (float*)d_out, nNodes, 4);
}
